// TriPixelMambaLayer_8615704396478
// MI455X (gfx1250) — matrix operand order repaired and checked by reading, not compiled, not run on hardware
//
#include <hip/hip_runtime.h>
#include <hip/hip_fp16.h>

typedef __attribute__((ext_vector_type(16))) _Float16 v16h;
typedef __attribute__((ext_vector_type(8)))  _Float16 v8h;
typedef __attribute__((ext_vector_type(8)))  float    v8f;

static constexpr int NBq   = 8;     // sub-batches (2^3 patches)
static constexpr int CD    = 64;    // model dim
static constexpr int LQ    = 4096;  // sequence length per sub-batch
static constexpr int DI    = 128;   // d_inner
static constexpr int E2    = 256;   // 2*d_inner
static constexpr int DST   = 16;    // d_state
static constexpr int NCH   = 64;    // scan chunks
static constexpr int CHL   = 64;    // chunk length (NCH*CHL == LQ)
static constexpr int EF    = 160;   // fused xproj cols: 128 dt + 16 B + 16 C

// ---- index helpers -------------------------------------------------------
__device__ __forceinline__ int dmap(int dir, int t) {
  if (dir == 0) return t;                      // forward
  if (dir == 1) return (LQ - 1) - t;           // backward
  return ((t & 15) << 8) | (t >> 4);           // slice permutation (n=16)
}
// (nb,l,c) -> flat index into x (1,64,32,32,32)
__device__ __forceinline__ int xidx(int nb, int l, int c) {
  int pz = (nb >> 2) & 1, ph = (nb >> 1) & 1, pw = nb & 1;
  int nz = l >> 8, nh = (l >> 4) & 15, nw = l & 15;
  return c * 32768 + (2 * nz + pz) * 1024 + (2 * nh + ph) * 32 + (2 * nw + pw);
}

// ---- async stage: global -> LDS via CDNA5 async-copy unit ----------------
// Each lane copies 16B memory->LDS; ASYNCcnt tracked; drained with
// s_wait_asynccnt then block barrier. LDS byte address = flat addr[31:0].
__device__ __forceinline__ void async_stage(const _Float16* __restrict__ g,
                                            _Float16* s, int halfs_total) {
  uint32_t base = (uint32_t)(uintptr_t)(void*)s;
  for (int i = threadIdx.x * 8; i < halfs_total; i += blockDim.x * 8) {
    uint32_t lds = base + (uint32_t)i * 2u;
    uint64_t ga  = (uint64_t)(uintptr_t)(g + i);
    asm volatile("global_load_async_to_lds_b128 %0, %1, off"
                 :: "v"(lds), "v"(ga) : "memory");
  }
  asm volatile("s_wait_asynccnt 0x0" ::: "memory");
  __syncthreads();
}

// ---- WMMA fragment loaders (16-bit, wave32) ------------------------------
// A (MxK): lane m=tl&15, half=tl>>4; elems 0..7 = K=k0+half*8+j, 8..15 = K=k0+16+half*8+j
__device__ __forceinline__ v16h load_fragA(const _Float16* __restrict__ base,
                                           int row, int ldk, int k0, int half) {
  const _Float16* p = base + (size_t)row * ldk + k0 + half * 8;
  v8h lo = *(const v8h*)(p);
  v8h hi = *(const v8h*)(p + 16);
  v16h r;
#pragma unroll
  for (int i = 0; i < 8; ++i) { r[i] = lo[i]; r[8 + i] = hi[i]; }
  return r;
}
// B (KxN) supplied as Bt row-major (N x K): lane n elems 0..7 = K=k0+half*8+j, 8..15 = K=k0+16+half*8+j
__device__ __forceinline__ v16h load_fragB(const _Float16* __restrict__ baseT,
                                           int col, int ldk, int k0, int half) {
  const _Float16* p = baseT + (size_t)col * ldk + k0 + half * 8;
  v8h lo = *(const v8h*)(p);
  v8h hi = *(const v8h*)(p + 16);
  v16h r;
#pragma unroll
  for (int i = 0; i < 8; ++i) { r[i] = lo[i]; r[8 + i] = hi[i]; }
  return r;
}
__device__ __forceinline__ v8f wmma16(v16h a, v16h b, v8f c) {
  return __builtin_amdgcn_wmma_f32_16x16x32_f16(false, a, false, b, (short)0, c,
                                                false, false);
}
__device__ __forceinline__ float siluf(float v) { return v / (1.f + __expf(-v)); }

// ---- kernels -------------------------------------------------------------
__global__ void k_prep_global(const float* __restrict__ inw, const float* __restrict__ outw,
                              _Float16* __restrict__ hin, _Float16* __restrict__ hout) {
  for (int i = blockIdx.x * blockDim.x + threadIdx.x; i < E2 * CD + CD * DI;
       i += blockDim.x * gridDim.x) {
    if (i < E2 * CD) hin[i] = (_Float16)inw[i];
    else             hout[i - E2 * CD] = (_Float16)outw[i - E2 * CD];
  }
}

__global__ void k_ln(const float* __restrict__ x, const float* __restrict__ g,
                     const float* __restrict__ b, _Float16* __restrict__ xn) {
  int idx = blockIdx.x * blockDim.x + threadIdx.x;   // nb*LQ
  int nb = idx >> 12, l = idx & (LQ - 1);
  float s = 0.f, ss = 0.f;
  for (int c = 0; c < CD; ++c) { float v = x[xidx(nb, l, c)]; s += v; ss += v * v; }
  float m = s * (1.f / CD);
  float inv = rsqrtf(ss * (1.f / CD) - m * m + 1e-5f);
  _Float16* o = xn + (size_t)idx * CD;
  for (int c = 0; c < CD; ++c)
    o[c] = (_Float16)((x[xidx(nb, l, c)] - m) * inv * g[c] + b[c]);
}

// xz[nb,e,l] = in_proj_w(256x64) @ xn^T.
// Block stages W (32KB) in LDS; each wave computes a 64(e) x 16(l) strip,
// reusing the xn B-fragment across 4 A tiles.
__global__ void k_inproj(const _Float16* __restrict__ hW, const _Float16* __restrict__ xn,
                         float* __restrict__ xz) {
  __shared__ _Float16 sW[E2 * CD];
  async_stage(hW, sW, E2 * CD);
  int wave = blockIdx.x * (blockDim.x >> 5) + (threadIdx.x >> 5);
  int tl = threadIdx.x & 31, half = tl >> 4, ln16 = tl & 15;
  int lt = wave & 255; wave >>= 8;
  int eg = wave & 3, nb = wave >> 2;
  int l0 = lt * 16;
  const _Float16* xb = xn + (size_t)(nb * LQ + l0) * CD;
  v8f acc[4] = {};
#pragma unroll
  for (int k0 = 0; k0 < CD; k0 += 32) {
    v16h b = load_fragB(xb, ln16, CD, k0, half);
#pragma unroll
    for (int e = 0; e < 4; ++e) {
      v16h a = load_fragA(sW, (eg * 4 + e) * 16 + ln16, CD, k0, half);
      acc[e] = wmma16(a, b, acc[e]);
    }
  }
#pragma unroll
  for (int e = 0; e < 4; ++e) {
    float* d = xz + ((size_t)nb * E2 + (eg * 4 + e) * 16) * LQ + l0;
#pragma unroll
    for (int r = 0; r < 8; ++r) d[(size_t)(r + 8 * half) * LQ + ln16] = acc[e][r];
  }
}

// fused weights: rows 0..127 = dt_w(128x4) @ xproj_w[0:4,:]; 128..143 = B rows; 144..159 = C rows
__global__ void k_prep_dir(const float* __restrict__ xw, const float* __restrict__ dw,
                           const float* __restrict__ Alog, _Float16* __restrict__ wf,
                           float* __restrict__ negA) {
  for (int i = threadIdx.x; i < EF * DI; i += blockDim.x) {
    int r = i >> 7, c = i & 127;
    float v;
    if (r < DI)
      v = dw[r * 4 + 0] * xw[0 * DI + c] + dw[r * 4 + 1] * xw[1 * DI + c] +
          dw[r * 4 + 2] * xw[2 * DI + c] + dw[r * 4 + 3] * xw[3 * DI + c];
    else if (r < DI + DST) v = xw[(4 + r - DI) * DI + c];
    else                   v = xw[(20 + r - DI - DST) * DI + c];
    wf[i] = (_Float16)v;
  }
  for (int i = threadIdx.x; i < DI * DST; i += blockDim.x)
    negA[i] = -__expf(Alog[i]);
}

// causal conv4 + SiLU, gathered along the direction's scan order
__global__ void k_conv(const float* __restrict__ xz, const float* __restrict__ cw,
                       const float* __restrict__ cb, _Float16* __restrict__ xi, int dir) {
  int idx = blockIdx.x * blockDim.x + threadIdx.x;
  int t = idx & (LQ - 1), ch = (idx >> 12) & 127, nb = idx >> 19;
  const float* row = xz + ((size_t)nb * E2 + ch) * LQ;
  float acc = cb[ch];
#pragma unroll
  for (int j = 0; j < 4; ++j) {
    int tt = t - 3 + j;
    if (tt >= 0) acc += cw[ch * 4 + j] * row[dmap(dir, tt)];
  }
  xi[(size_t)(nb * LQ + t) * DI + ch] = (_Float16)siluf(acc);
}

// dbc[nb,t,0:160]: dt (softplus) | B | C.  Block stages fused W (40KB) in LDS;
// each wave computes a full 16(t) x 160 strip (10 accumulators), A loaded once
// per k-step and reused across all 10 N tiles.
__global__ void k_xproj(const _Float16* __restrict__ xi, const _Float16* __restrict__ wf,
                        const float* __restrict__ dtb, float* __restrict__ dbc) {
  __shared__ _Float16 sW[EF * DI];
  async_stage(wf, sW, EF * DI);
  int wave = blockIdx.x * (blockDim.x >> 5) + (threadIdx.x >> 5);
  int tl = threadIdx.x & 31, half = tl >> 4, ln16 = tl & 15;
  int lt = wave & 255, nb = wave >> 8;
  int t0 = lt * 16;
  const _Float16* ab = xi + (size_t)(nb * LQ + t0) * DI;
  v8f acc[10] = {};
#pragma unroll
  for (int k0 = 0; k0 < DI; k0 += 32) {
    v16h a = load_fragA(ab, ln16, DI, k0, half);
#pragma unroll
    for (int n = 0; n < 10; ++n) {
      v16h b = load_fragB(sW, n * 16 + ln16, DI, k0, half);
      acc[n] = wmma16(a, b, acc[n]);
    }
  }
#pragma unroll
  for (int n = 0; n < 10; ++n) {
    int col = n * 16 + ln16;
    float bias = (col < DI) ? dtb[col] : 0.f;
#pragma unroll
    for (int r = 0; r < 8; ++r) {
      float v = acc[n][r];
      if (col < DI) { v += bias; v = (v > 20.f) ? v : __logf(1.f + __expf(v)); }
      dbc[(size_t)(nb * LQ + t0 + r + 8 * half) * EF + col] = v;
    }
  }
}

// phase 1: per-(nb,d,chunk) 16-state chunk summaries (aProd, hEnd)
__global__ void k_scan1(const float* __restrict__ dbc, const _Float16* __restrict__ xi,
                        const float* __restrict__ negA, float* __restrict__ summ) {
  int idx = blockIdx.x * blockDim.x + threadIdx.x;   // nb(3) d(7) c(6)
  int c = idx & 63, d = (idx >> 6) & 127, nb = idx >> 13;
  float h[DST], aP[DST];
#pragma unroll
  for (int s = 0; s < DST; ++s) { h[s] = 0.f; aP[s] = 1.f; }
  const float* nA = negA + d * DST;
  for (int i = 0; i < CHL; ++i) {
    int t = c * CHL + i;
    const float* row = dbc + (size_t)(nb * LQ + t) * EF;
    float dt = row[d];
    float dx = dt * (float)xi[(size_t)(nb * LQ + t) * DI + d];
#pragma unroll
    for (int s = 0; s < DST; ++s) {
      float a = __expf(dt * nA[s]);
      h[s] = fmaf(a, h[s], dx * row[DI + s]);
      aP[s] *= a;
    }
  }
#pragma unroll
  for (int s = 0; s < DST; ++s) {
    size_t i2 = ((size_t)((nb * DI + d) * DST + s)) * NCH + c;
    summ[i2 * 2 + 0] = aP[s];
    summ[i2 * 2 + 1] = h[s];
  }
}

// phase 2: exclusive scan over chunk summaries per (nb,d,s)
__global__ void k_scan2(const float* __restrict__ summ, float* __restrict__ carry) {
  int ch = blockIdx.x * blockDim.x + threadIdx.x;    // 8*128*16
  float h0 = 0.f;
  for (int c = 0; c < NCH; ++c) {
    size_t i = (size_t)ch * NCH + c;
    carry[i] = h0;
    h0 = summ[i * 2] * h0 + summ[i * 2 + 1];
  }
}

// phase 3: recompute with carry, contract with C, +xi*D, SiLU(z) gate, accumulate
__global__ void k_scan3(const float* __restrict__ dbc, const _Float16* __restrict__ xi,
                        const float* __restrict__ negA, const float* __restrict__ carry,
                        const float* __restrict__ Dp, const float* __restrict__ xz,
                        float* __restrict__ oacc, int dir) {
  int idx = blockIdx.x * blockDim.x + threadIdx.x;
  int c = idx & 63, d = (idx >> 6) & 127, nb = idx >> 13;
  float h[DST];
#pragma unroll
  for (int s = 0; s < DST; ++s)
    h[s] = carry[((size_t)((nb * DI + d) * DST + s)) * NCH + c];
  const float* nA = negA + d * DST;
  const float* zrow = xz + ((size_t)nb * E2 + DI + d) * LQ;
  float Dd = Dp[d];
  for (int i = 0; i < CHL; ++i) {
    int t = c * CHL + i;
    const float* row = dbc + (size_t)(nb * LQ + t) * EF;
    float dt = row[d];
    float xiv = (float)xi[(size_t)(nb * LQ + t) * DI + d];
    float dx = dt * xiv;
    float ysum = 0.f;
#pragma unroll
    for (int s = 0; s < DST; ++s) {
      float a = __expf(dt * nA[s]);
      h[s] = fmaf(a, h[s], dx * row[DI + s]);
      ysum = fmaf(h[s], row[DI + DST + s], ysum);
    }
    float y = ysum + xiv * Dd;
    int l = dmap(dir, t);
    y *= siluf(zrow[l]);
    oacc[(size_t)(nb * LQ + l) * DI + d] += y;
  }
}

__global__ void k_cvt(const float* __restrict__ a, _Float16* __restrict__ b) {
  int i = blockIdx.x * blockDim.x + threadIdx.x;
  b[i] = (_Float16)a[i];
}

// out_proj + un-patch scatter + residual. Block stages Wout (16KB) in LDS;
// each wave computes a 16(l) x 64(c) strip (A reused across 4 c tiles).
__global__ void k_outproj(const _Float16* __restrict__ y, const _Float16* __restrict__ hW,
                          const float* __restrict__ x, float* __restrict__ out) {
  __shared__ _Float16 sW[CD * DI];
  async_stage(hW, sW, CD * DI);
  int wave = blockIdx.x * (blockDim.x >> 5) + (threadIdx.x >> 5);
  int tl = threadIdx.x & 31, half = tl >> 4, ln16 = tl & 15;
  int lt = wave & 255, nb = wave >> 8;
  int l0 = lt * 16;
  const _Float16* ab = y + (size_t)(nb * LQ + l0) * DI;
  v8f acc[4] = {};
#pragma unroll
  for (int k0 = 0; k0 < DI; k0 += 32) {
    v16h a = load_fragA(ab, ln16, DI, k0, half);
#pragma unroll
    for (int ct = 0; ct < 4; ++ct) {
      v16h b = load_fragB(sW, ct * 16 + ln16, DI, k0, half);
      acc[ct] = wmma16(a, b, acc[ct]);
    }
  }
#pragma unroll
  for (int ct = 0; ct < 4; ++ct) {
    int cch = ct * 16 + ln16;
#pragma unroll
    for (int r = 0; r < 8; ++r) {
      int l = l0 + r + 8 * half;
      int gi = xidx(nb, l, cch);
      out[gi] = x[gi] + acc[ct][r];
    }
  }
}

// ---- launch --------------------------------------------------------------
extern "C" void kernel_launch(void* const* d_in, const int* in_sizes, int n_in,
                              void* d_out, int out_size, void* d_ws, size_t ws_size,
                              hipStream_t stream) {
  (void)in_sizes; (void)n_in; (void)out_size; (void)ws_size;
  const float* x    = (const float*)d_in[0];
  const float* ng   = (const float*)d_in[1];
  const float* nb_  = (const float*)d_in[2];
  const float* inw  = (const float*)d_in[3];
  const float* outw = (const float*)d_in[4];

  char* w = (char*)d_ws;
  size_t o = 0;
  auto alloc = [&](size_t bytes) -> void* {
    void* p = w + o; o += (bytes + 255) & ~(size_t)255; return p;
  };
  _Float16* xn    = (_Float16*)alloc((size_t)NBq * LQ * CD * 2);
  _Float16* hin   = (_Float16*)alloc((size_t)E2 * CD * 2);
  _Float16* hout  = (_Float16*)alloc((size_t)CD * DI * 2);
  float*    xz    = (float*)   alloc((size_t)NBq * E2 * LQ * 4);
  float*    oacc  = (float*)   alloc((size_t)NBq * LQ * DI * 4);
  _Float16* yh    = (_Float16*)alloc((size_t)NBq * LQ * DI * 2);
  _Float16* xi    = (_Float16*)alloc((size_t)NBq * LQ * DI * 2);
  _Float16* wf    = (_Float16*)alloc((size_t)EF * DI * 2);
  float*    negA  = (float*)   alloc((size_t)DI * DST * 4);
  float*    dbc   = (float*)   alloc((size_t)NBq * LQ * EF * 4);
  float*    summ  = (float*)   alloc((size_t)NBq * DI * DST * NCH * 2 * 4);
  float*    carry = (float*)   alloc((size_t)NBq * DI * DST * NCH * 4);

  hipMemsetAsync(oacc, 0, (size_t)NBq * LQ * DI * 4, stream);
  k_prep_global<<<64, 256, 0, stream>>>(inw, outw, hin, hout);
  k_ln<<<(NBq * LQ) / 256, 256, 0, stream>>>(x, ng, nb_, xn);
  k_inproj<<<(NBq * 4 * (LQ / 16)) / 8, 256, 0, stream>>>(hin, xn, xz);

  for (int dir = 0; dir < 3; ++dir) {
    const float* cw   = (const float*)d_in[5 + 7 * dir];
    const float* cb   = (const float*)d_in[6 + 7 * dir];
    const float* xw   = (const float*)d_in[7 + 7 * dir];
    const float* dw   = (const float*)d_in[8 + 7 * dir];
    const float* dtb  = (const float*)d_in[9 + 7 * dir];
    const float* Alog = (const float*)d_in[10 + 7 * dir];
    const float* Dp   = (const float*)d_in[11 + 7 * dir];

    k_prep_dir<<<1, 256, 0, stream>>>(xw, dw, Alog, wf, negA);
    k_conv<<<(NBq * DI * LQ) / 256, 256, 0, stream>>>(xz, cw, cb, xi, dir);
    k_xproj<<<(NBq * (LQ / 16)) / 8, 256, 0, stream>>>(xi, wf, dtb, dbc);
    k_scan1<<<(NBq * DI * NCH) / 256, 256, 0, stream>>>(dbc, xi, negA, summ);
    k_scan2<<<(NBq * DI * DST) / 256, 256, 0, stream>>>(summ, carry);
    k_scan3<<<(NBq * DI * NCH) / 256, 256, 0, stream>>>(dbc, xi, negA, carry, Dp, xz,
                                                        oacc, dir);
  }
  k_cvt<<<(NBq * LQ * DI) / 256, 256, 0, stream>>>(oacc, yh);
  k_outproj<<<(NBq * (LQ / 16)) / 8, 256, 0, stream>>>(yh, hout, x, (float*)d_out);
}
